// hwnetSynthSPP_25804163514863
// MI455X (gfx1250) — compile-verified, matrix operand order checked by reading, not run on hardware
//
#include <hip/hip_runtime.h>
#include <hip/hip_bf16.h>

// ---------------------------------------------------------------------------
// CDNA5 (gfx1250) implementation: WMMA f32_16x16x32_f16 for all GEMM-like
// stages (implicit-GEMM convs + FC layers), scalar kernels for BN/pool/SPP.
// wave32; 256-thread blocks = 8 waves; 128x128 output tile per block;
// each wave computes 32x64 -> 8 v_wmma per K-step (0.75 LDS frag loads/WMMA).
// Workspace requirement: (12582912 + 25165824 + 8192) floats ~= 151 MB.
// ---------------------------------------------------------------------------

typedef __attribute__((ext_vector_type(16))) _Float16 v16h;
typedef __attribute__((ext_vector_type(8)))  _Float16 v8h;
typedef __attribute__((ext_vector_type(8)))  float    v8f;

#define BN_EPS 1e-5f
#define NEGV  -1e30f

__device__ __forceinline__ v16h join8(v8h lo, v8h hi) {
    v16h r;
#pragma unroll
    for (int i = 0; i < 8; ++i) { r[i] = lo[i]; r[8 + i] = hi[i]; }
    return r;
}

// ---------------------------------------------------------------------------
// Implicit-GEMM convolution:
//   out[b][m][oy][ox] = bias[m] + sum_{cin,dy,dx} in[b][cin][oy+dy-P][ox+dx-P]
//                                                 * wgt[m][cin][dy][dx]
// GEMM view: M = Cout (weights OIHW row-major = M x K), N = B*Hout*Wout,
//            K = Cin*KH*KW.  Tile: 128(M) x 128(N), K-step 32.
// 8 waves as 4(M) x 2(N); per wave: 2 A frags x 4 B frags -> 8 wmma / K-step.
// ---------------------------------------------------------------------------
__global__ __launch_bounds__(256)
void conv_wmma_kernel(const float* __restrict__ in, const float* __restrict__ wgt,
                      const float* __restrict__ bias, float* __restrict__ out,
                      int Bn, int Cin, int Hin, int Win,
                      int Cout, int Hout, int Wout,
                      int KH, int KW, int PAD)
{
    const int HWo  = Hout * Wout;
    const int KK   = KH * KW;
    const int Ktot = Cin * KK;
    const int n0   = blockIdx.x * 128;
    const int m0   = blockIdx.y * 128;
    const int tid  = threadIdx.x;
    const int lane = tid & 31;
    const int wave = tid >> 5;
    const int wm   = wave >> 1;   // 0..3 -> 32-row M strip
    const int wn   = wave & 1;    // 0..1 -> 64-col N strip

    __shared__ __align__(16) _Float16 As[128][40];   // 128 rows x 32 K (+8 pad)
    __shared__ __align__(16) _Float16 Bs[128][40];   // 128 cols x 32 K (+8 pad)

    v8f acc[2][4];
#pragma unroll
    for (int s = 0; s < 2; ++s)
#pragma unroll
        for (int j = 0; j < 4; ++j)
#pragma unroll
            for (int i = 0; i < 8; ++i) acc[s][j][i] = 0.0f;

    // cooperative-fill coordinates (both tiles are 128 x 32 halves)
    const int frow = tid >> 1;            // 128 rows/cols
    const int fkk  = (tid & 1) * 16;      // 2 x 16 halves

    // im2col source pixel for the B tile (fixed per thread)
    const int bn   = n0 + frow;
    const int bb   = bn / HWo;
    const int brem = bn - bb * HWo;
    const int boy  = brem / Wout;
    const int box  = brem - boy * Wout;

    for (int k0 = 0; k0 < Ktot; k0 += 32) {
        const bool fullk = (k0 + 32 <= Ktot);
        // ---- A tile: weights (row-major M x K), f32 -> f16 ----
        {
            const bool rowok = (m0 + frow) < Cout;
            const float* wrow = wgt + (size_t)(m0 + frow) * Ktot;
            if (fullk && rowok) {
                if (k0 + 64 < Ktot) __builtin_prefetch(wrow + k0 + 64, 0, 3);
#pragma unroll
                for (int i = 0; i < 16; ++i)
                    As[frow][fkk + i] = (_Float16)wrow[k0 + fkk + i];
            } else {
#pragma unroll 4
                for (int i = 0; i < 16; ++i) {
                    int k = k0 + fkk + i;
                    As[frow][fkk + i] =
                        (rowok && k < Ktot) ? (_Float16)wrow[k] : (_Float16)0.0f;
                }
            }
        }
        // ---- B tile: on-the-fly im2col gather with incremental (cin,dy,dx) ----
        {
            int k   = k0 + fkk;
            int cin = k / KK;
            int r2  = k - cin * KK;
            int dy  = r2 / KW;
            int dx  = r2 - dy * KW;
#pragma unroll 4
            for (int i = 0; i < 16; ++i, ++k) {
                float v = 0.0f;
                if (k < Ktot) {
                    int iy = boy + dy - PAD;
                    int ix = box + dx - PAD;
                    if (iy >= 0 && iy < Hin && ix >= 0 && ix < Win)
                        v = in[(((size_t)bb * Cin + cin) * Hin + iy) * Win + ix];
                }
                Bs[frow][fkk + i] = (_Float16)v;
                // odometer: advance (dx,dy,cin)
                ++dx;
                if (dx == KW) { dx = 0; ++dy; if (dy == KH) { dy = 0; ++cin; } }
            }
        }
        __syncthreads();

        // ---- fragments per ISA 16-bit layout: lane L holds row/col L%16,
        //      K chunks {8*(L>=16) .. +7} and {+16 .. +23} ----
        const int fr = lane & 15;
        const int kp = (lane >> 4) * 8;
        v16h afrag[2];
#pragma unroll
        for (int s = 0; s < 2; ++s)
            afrag[s] = join8(*(const v8h*)&As[wm * 32 + s * 16 + fr][kp],
                             *(const v8h*)&As[wm * 32 + s * 16 + fr][kp + 16]);
#pragma unroll
        for (int j = 0; j < 4; ++j) {
            v16h bfrag = join8(*(const v8h*)&Bs[wn * 64 + j * 16 + fr][kp],
                               *(const v8h*)&Bs[wn * 64 + j * 16 + fr][kp + 16]);
#pragma unroll
            for (int s = 0; s < 2; ++s)
                acc[s][j] = __builtin_amdgcn_wmma_f32_16x16x32_f16(
                    false, afrag[s], false, bfrag, (short)0, acc[s][j], false, false);
        }
        __syncthreads();
    }

    // ---- store: C layout VGPR t -> M = t + 8*(lane>=16), N = lane%16 ----
    const int fr = lane & 15;
    const int hi = lane >> 4;
#pragma unroll
    for (int s = 0; s < 2; ++s) {
#pragma unroll
        for (int j = 0; j < 4; ++j) {
            int n   = n0 + wn * 64 + j * 16 + fr;
            int b   = n / HWo;
            int rem = n - b * HWo;
            int oy  = rem / Wout;
            int ox  = rem - oy * Wout;
#pragma unroll
            for (int t = 0; t < 8; ++t) {
                int m = m0 + wm * 32 + s * 16 + t + 8 * hi;
                if (m < Cout)
                    out[(((size_t)b * Cout + m) * Hout + oy) * Wout + ox] =
                        acc[s][j][t] + bias[m];
            }
        }
    }
}

// ---------------------------------------------------------------------------
// FC GEMM: out(MxN) = A(MxK, row-major) * Wt(NxK, row-major)^T + bias[n]
// Same 128x128 WMMA core as the conv kernel.
// ---------------------------------------------------------------------------
__global__ __launch_bounds__(256)
void fc_wmma_kernel(const float* __restrict__ A, const float* __restrict__ Wt,
                    const float* __restrict__ bias, float* __restrict__ out,
                    int M, int N, int K)
{
    const int n0   = blockIdx.x * 128;
    const int m0   = blockIdx.y * 128;
    const int tid  = threadIdx.x;
    const int lane = tid & 31;
    const int wave = tid >> 5;
    const int wm   = wave >> 1;
    const int wn   = wave & 1;

    __shared__ __align__(16) _Float16 As[128][40];
    __shared__ __align__(16) _Float16 Bs[128][40];

    v8f acc[2][4];
#pragma unroll
    for (int s = 0; s < 2; ++s)
#pragma unroll
        for (int j = 0; j < 4; ++j)
#pragma unroll
            for (int i = 0; i < 8; ++i) acc[s][j][i] = 0.0f;

    const int frow = tid >> 1;
    const int fkk  = (tid & 1) * 16;
    const float* arow = A  + (size_t)(m0 + frow) * K;
    const float* wrow = Wt + (size_t)(n0 + frow) * K;
    const bool  arok  = (m0 + frow) < M;

    for (int k0 = 0; k0 < K; k0 += 32) {
        const bool fullk = (k0 + 32 <= K);
        if (fullk && arok) {
#pragma unroll
            for (int i = 0; i < 16; ++i)
                As[frow][fkk + i] = (_Float16)arow[k0 + fkk + i];
        } else {
#pragma unroll 4
            for (int i = 0; i < 16; ++i) {
                int k = k0 + fkk + i;
                As[frow][fkk + i] = (arok && k < K) ? (_Float16)arow[k] : (_Float16)0.0f;
            }
        }
        if (fullk) {
            if (k0 + 64 < K) __builtin_prefetch(wrow + k0 + 64, 0, 3);
#pragma unroll
            for (int i = 0; i < 16; ++i)
                Bs[frow][fkk + i] = (_Float16)wrow[k0 + fkk + i];
        } else {
#pragma unroll 4
            for (int i = 0; i < 16; ++i) {
                int k = k0 + fkk + i;
                Bs[frow][fkk + i] = (k < K) ? (_Float16)wrow[k] : (_Float16)0.0f;
            }
        }
        __syncthreads();

        const int fr = lane & 15;
        const int kp = (lane >> 4) * 8;
        v16h afrag[2];
#pragma unroll
        for (int s = 0; s < 2; ++s)
            afrag[s] = join8(*(const v8h*)&As[wm * 32 + s * 16 + fr][kp],
                             *(const v8h*)&As[wm * 32 + s * 16 + fr][kp + 16]);
#pragma unroll
        for (int j = 0; j < 4; ++j) {
            v16h bfrag = join8(*(const v8h*)&Bs[wn * 64 + j * 16 + fr][kp],
                               *(const v8h*)&Bs[wn * 64 + j * 16 + fr][kp + 16]);
#pragma unroll
            for (int s = 0; s < 2; ++s)
                acc[s][j] = __builtin_amdgcn_wmma_f32_16x16x32_f16(
                    false, afrag[s], false, bfrag, (short)0, acc[s][j], false, false);
        }
        __syncthreads();
    }

    const int fr = lane & 15;
    const int hi = lane >> 4;
#pragma unroll
    for (int s = 0; s < 2; ++s) {
#pragma unroll
        for (int j = 0; j < 4; ++j) {
            int n = n0 + wn * 64 + j * 16 + fr;
#pragma unroll
            for (int t = 0; t < 8; ++t) {
                int m = m0 + wm * 32 + s * 16 + t + 8 * hi;
                if (m < M)
                    out[(size_t)m * N + n] = acc[s][j][t] + bias[n];
            }
        }
    }
}

// ---------------------------------------------------------------------------
// Elementwise / reduction helper kernels
// ---------------------------------------------------------------------------
__global__ void pad_edge_kernel(const float* __restrict__ x, float* __restrict__ xp,
                                int Bn, int H, int Wd, int P)
{
    const int Hp = H + 2 * P, Wp = Wd + 2 * P;
    const int total = Bn * Hp * Wp;
    for (int i = blockIdx.x * blockDim.x + threadIdx.x; i < total;
         i += gridDim.x * blockDim.x) {
        int b  = i / (Hp * Wp);
        int r  = i - b * (Hp * Wp);
        int yy = r / Wp;
        int xx = r - yy * Wp;
        int sy = min(max(yy - P, 0), H - 1);
        int sx = min(max(xx - P, 0), Wd - 1);
        xp[i] = x[((size_t)b * H + sy) * Wd + sx];
    }
}

__global__ __launch_bounds__(256)
void bn2d_stats_kernel(const float* __restrict__ x, float* __restrict__ mean,
                       float* __restrict__ rstd, int Bn, int C, int HW)
{
    const int c = blockIdx.x;
    const int cnt = Bn * HW;
    float s = 0.0f, sq = 0.0f;
    for (int i = threadIdx.x; i < cnt; i += 256) {
        int b = i / HW;
        int p = i - b * HW;
        float v = x[((size_t)b * C + c) * HW + p];
        s += v; sq += v * v;
    }
    __shared__ float ss[256], sqq[256];
    ss[threadIdx.x] = s; sqq[threadIdx.x] = sq;
    __syncthreads();
    for (int off = 128; off > 0; off >>= 1) {
        if (threadIdx.x < off) {
            ss[threadIdx.x]  += ss[threadIdx.x + off];
            sqq[threadIdx.x] += sqq[threadIdx.x + off];
        }
        __syncthreads();
    }
    if (threadIdx.x == 0) {
        float m = ss[0] / (float)cnt;
        float v = sqq[0] / (float)cnt - m * m;
        mean[c] = m;
        rstd[c] = rsqrtf(v + BN_EPS);
    }
}

__global__ void bn2d_relu_kernel(float* __restrict__ x, const float* __restrict__ mean,
                                 const float* __restrict__ rstd, const float* __restrict__ g,
                                 const float* __restrict__ b, int C, int HW, int total)
{
    for (int i = blockIdx.x * blockDim.x + threadIdx.x; i < total;
         i += gridDim.x * blockDim.x) {
        int c = (i / HW) % C;
        float v = (x[i] - mean[c]) * rstd[c] * g[c] + b[c];
        x[i] = v > 0.0f ? v : 0.0f;
    }
}

__global__ void maxpool2_kernel(const float* __restrict__ x, float* __restrict__ y,
                                int BC, int H, int Wd)
{
    const int Ho = H >> 1, Wo = Wd >> 1;
    const int total = BC * Ho * Wo;
    for (int i = blockIdx.x * blockDim.x + threadIdx.x; i < total;
         i += gridDim.x * blockDim.x) {
        int bc = i / (Ho * Wo);
        int r  = i - bc * (Ho * Wo);
        int oy = r / Wo;
        int ox = r - oy * Wo;
        const float* p = x + ((size_t)bc * H + oy * 2) * Wd + ox * 2;
        y[i] = fmaxf(fmaxf(p[0], p[1]), fmaxf(p[Wd], p[Wd + 1]));
    }
}

// one thread per (roi, channel): column-max over roi rows, then strip maxes
__global__ void roi_spp_kernel(const float* __restrict__ feat, const int* __restrict__ roi,
                               float* __restrict__ out, int R, int C, int Hf, int Wf)
{
    const int idx = blockIdx.x * blockDim.x + threadIdx.x;
    if (idx >= R * C) return;
    const int r = idx / C;
    const int c = idx - r * C;
    const int im = roi[r * 5 + 0];
    const int c1 = (int)((float)roi[r * 5 + 1] * 0.125f);
    const int r1 = (int)((float)roi[r * 5 + 2] * 0.125f);
    const int c2 = (int)((float)roi[r * 5 + 3] * 0.125f);
    const int r2 = (int)((float)roi[r * 5 + 4] * 0.125f);
    const float* f = feat + ((size_t)im * C + c) * Hf * Wf;

    float colmax[16];
#pragma unroll
    for (int j = 0; j < 16; ++j) colmax[j] = NEGV;
    const int rlo = max(r1, 0), rhi = min(r2, Hf - 1);
    for (int y = rlo; y <= rhi; ++y)
        for (int j = 0; j < Wf; ++j)
            colmax[j] = fmaxf(colmax[j], f[y * Wf + j]);

    const int w = c2 - c1 + 1;
    int off = 0;
#pragma unroll
    for (int l = 1; l <= 3; ++l) {
        for (int k = 0; k < l; ++k) {
            int lo = c1 + (k * w) / l;
            int hi = c1 + ((k + 1) * w + l - 1) / l - 1;
            float m = NEGV;
            int jlo = max(lo, 0), jhi = min(hi, Wf - 1);
            for (int j = jlo; j <= jhi; ++j) m = fmaxf(m, colmax[j]);
            out[(size_t)r * (6 * C) + off + c * l + k] = m;
        }
        off += C * l;
    }
}

__global__ void bn1d_stats_kernel(const float* __restrict__ x, float* __restrict__ mean,
                                  float* __restrict__ rstd, int M, int F)
{
    const int f = blockIdx.x * blockDim.x + threadIdx.x;
    if (f >= F) return;
    float s = 0.0f, sq = 0.0f;
    for (int r = 0; r < M; ++r) {
        float v = x[(size_t)r * F + f];
        s += v; sq += v * v;
    }
    float m = s / (float)M;
    float v = sq / (float)M - m * m;
    mean[f] = m;
    rstd[f] = rsqrtf(v + BN_EPS);
}

__global__ void bn1d_relu_kernel(float* __restrict__ x, const float* __restrict__ mean,
                                 const float* __restrict__ rstd, const float* __restrict__ g,
                                 const float* __restrict__ b, int M, int F)
{
    const int i = blockIdx.x * blockDim.x + threadIdx.x;
    if (i >= M * F) return;
    const int f = i % F;
    float v = (x[i] - mean[f]) * rstd[f] * g[f] + b[f];
    x[i] = v > 0.0f ? v : 0.0f;
}

__global__ void concat_kernel(const float* __restrict__ a, const float* __restrict__ b,
                              float* __restrict__ out, int M, int Fa, int Fb)
{
    const int F = Fa + Fb;
    const int total = M * F;
    for (int i = blockIdx.x * blockDim.x + threadIdx.x; i < total;
         i += gridDim.x * blockDim.x) {
        int r = i / F;
        int f = i - r * F;
        out[i] = (f < Fa) ? a[(size_t)r * Fa + f] : b[(size_t)r * Fb + (f - Fa)];
    }
}

// ---------------------------------------------------------------------------
// Host orchestration
// ---------------------------------------------------------------------------
extern "C" void kernel_launch(void* const* d_in, const int* in_sizes, int n_in,
                              void* d_out, int out_size, void* d_ws, size_t ws_size,
                              hipStream_t stream)
{
    const float* x    = (const float*)d_in[0];
    const int*   roi  = (const int*)  d_in[1];
    const float* phoc = (const float*)d_in[2];
    const float* c1w = (const float*)d_in[3],  *c1b = (const float*)d_in[4];
    const float* g1  = (const float*)d_in[5],  *bb1 = (const float*)d_in[6];
    const float* c2w = (const float*)d_in[7],  *c2b = (const float*)d_in[8];
    const float* g2  = (const float*)d_in[9],  *bb2 = (const float*)d_in[10];
    const float* c3w = (const float*)d_in[11], *c3b = (const float*)d_in[12];
    const float* g3  = (const float*)d_in[13], *bb3 = (const float*)d_in[14];
    const float* c4w = (const float*)d_in[15], *c4b = (const float*)d_in[16];
    const float* g4  = (const float*)d_in[17], *bb4 = (const float*)d_in[18];
    const float* c5w = (const float*)d_in[19], *c5b = (const float*)d_in[20];
    const float* g5  = (const float*)d_in[21], *bb5 = (const float*)d_in[22];
    const float* f1w = (const float*)d_in[23], *f1b = (const float*)d_in[24];
    const float* g6  = (const float*)d_in[25], *bb6 = (const float*)d_in[26];
    const float* f2w = (const float*)d_in[27], *f2b = (const float*)d_in[28];
    const float* g7  = (const float*)d_in[29], *bb7 = (const float*)d_in[30];
    float* out = (float*)d_out;

    // workspace: two ping-pong slabs + stats
    float* W = (float*)d_ws;
    const size_t S0 = 0;                 // slab0: 12,582,912 floats
    const size_t S1 = 12582912;          // slab1: 25,165,824 floats
    const size_t ST = S1 + 25165824;     // stats
    float* mean = W + ST;
    float* rstd = W + ST + 4096;

    float* xp   = W + S0;   // (64,1,52,132)
    float* a1   = W + S1;   // (64,64,48,128)
    float* p1   = W + S0;   // (64,64,24,64)
    float* a2   = W + S1;   // (64,128,24,64)
    float* p2   = W + S0;   // (64,128,12,32)
    float* a3   = W + S1;   // (64,256,12,32)
    float* a4   = W + S0;   // (64,512,12,32)
    float* p4   = W + S1;   // (64,512,6,16)
    float* a5   = W + S0;   // (64,512,6,16)
    float* spp  = W + S1;   // (256,3072)
    float* fc1o = W + S0;   // (256,2048)
    float* cat  = W + S1;   // (256,7901)

    const int EW = 4096;    // elementwise grid blocks (grid-stride)

    // stage 0: edge pad (64,1,48,128) -> (64,1,52,132)
    pad_edge_kernel<<<(64 * 52 * 132 + 255) / 256, 256, 0, stream>>>(x, xp, 64, 48, 128, 2);

    // conv1: K=25 (padded to 32), N=393216, M=64 (tile-M guarded)
    conv_wmma_kernel<<<dim3(3072, 1), 256, 0, stream>>>(
        xp, c1w, c1b, a1, 64, 1, 52, 132, 64, 48, 128, 5, 5, 0);
    bn2d_stats_kernel<<<64, 256, 0, stream>>>(a1, mean, rstd, 64, 64, 48 * 128);
    bn2d_relu_kernel<<<EW, 256, 0, stream>>>(a1, mean, rstd, g1, bb1, 64, 48 * 128, 64 * 64 * 48 * 128);
    maxpool2_kernel<<<EW, 256, 0, stream>>>(a1, p1, 64 * 64, 48, 128);

    // conv2: K=1600, N=98304, M=128
    conv_wmma_kernel<<<dim3(768, 1), 256, 0, stream>>>(
        p1, c2w, c2b, a2, 64, 64, 24, 64, 128, 24, 64, 5, 5, 2);
    bn2d_stats_kernel<<<128, 256, 0, stream>>>(a2, mean, rstd, 64, 128, 24 * 64);
    bn2d_relu_kernel<<<EW, 256, 0, stream>>>(a2, mean, rstd, g2, bb2, 128, 24 * 64, 64 * 128 * 24 * 64);
    maxpool2_kernel<<<EW, 256, 0, stream>>>(a2, p2, 64 * 128, 24, 64);

    // conv3: K=1152, N=24576, M=256
    conv_wmma_kernel<<<dim3(192, 2), 256, 0, stream>>>(
        p2, c3w, c3b, a3, 64, 128, 12, 32, 256, 12, 32, 3, 3, 1);
    bn2d_stats_kernel<<<256, 256, 0, stream>>>(a3, mean, rstd, 64, 256, 12 * 32);
    bn2d_relu_kernel<<<EW, 256, 0, stream>>>(a3, mean, rstd, g3, bb3, 256, 12 * 32, 64 * 256 * 12 * 32);

    // conv4: K=2304, N=24576, M=512
    conv_wmma_kernel<<<dim3(192, 4), 256, 0, stream>>>(
        a3, c4w, c4b, a4, 64, 256, 12, 32, 512, 12, 32, 3, 3, 1);
    bn2d_stats_kernel<<<512, 256, 0, stream>>>(a4, mean, rstd, 64, 512, 12 * 32);
    bn2d_relu_kernel<<<EW, 256, 0, stream>>>(a4, mean, rstd, g4, bb4, 512, 12 * 32, 64 * 512 * 12 * 32);
    maxpool2_kernel<<<EW, 256, 0, stream>>>(a4, p4, 64 * 512, 12, 32);

    // conv5: K=4608, N=6144, M=512
    conv_wmma_kernel<<<dim3(48, 4), 256, 0, stream>>>(
        p4, c5w, c5b, a5, 64, 512, 6, 16, 512, 6, 16, 3, 3, 1);
    bn2d_stats_kernel<<<512, 256, 0, stream>>>(a5, mean, rstd, 64, 512, 6 * 16);
    bn2d_relu_kernel<<<EW, 256, 0, stream>>>(a5, mean, rstd, g5, bb5, 512, 6 * 16, 64 * 512 * 6 * 16);

    // ROI SPP: (256 rois, 512 ch) -> (256, 3072)
    roi_spp_kernel<<<(256 * 512 + 255) / 256, 256, 0, stream>>>(a5, roi, spp, 256, 512, 6, 16);

    // fc1: (256x3072) @ (2048x3072)^T
    fc_wmma_kernel<<<dim3(16, 2), 256, 0, stream>>>(spp, f1w, f1b, fc1o, 256, 2048, 3072);
    bn1d_stats_kernel<<<8, 256, 0, stream>>>(fc1o, mean, rstd, 256, 2048);
    bn1d_relu_kernel<<<(256 * 2048 + 255) / 256, 256, 0, stream>>>(fc1o, mean, rstd, g6, bb6, 256, 2048);

    // concat (256,2048)+(256,5853) -> (256,7901)
    concat_kernel<<<EW, 256, 0, stream>>>(fc1o, phoc, cat, 256, 2048, 5853);

    // fc2: (256x7901) @ (2048x7901)^T  (K guarded, not a multiple of 32)
    fc_wmma_kernel<<<dim3(16, 2), 256, 0, stream>>>(cat, f2w, f2b, out, 256, 2048, 7901);
    bn1d_stats_kernel<<<8, 256, 0, stream>>>(out, mean, rstd, 256, 2048);
    bn1d_relu_kernel<<<(256 * 2048 + 255) / 256, 256, 0, stream>>>(out, mean, rstd, g7, bb7, 256, 2048);
}
